// CausalSelfAttention_46127948759423
// MI455X (gfx1250) — matrix operand order repaired and checked by reading, not compiled, not run on hardware
//
#include <hip/hip_runtime.h>
#include <hip/hip_bf16.h>

typedef __attribute__((ext_vector_type(16))) _Float16 v16h;
typedef __attribute__((ext_vector_type(8)))  _Float16 v8h;
typedef __attribute__((ext_vector_type(8)))  float    v8f;

static constexpr int B_   = 4;
static constexpr int T_   = 2048;
static constexpr int C_   = 1024;
static constexpr int H_   = 16;
static constexpr int DK_  = 64;

__device__ __forceinline__ v8f vzero8f() {
  v8f v;
#pragma unroll
  for (int i = 0; i < 8; ++i) v[i] = 0.0f;
  return v;
}

// ---- Fragment loaders (16-bit WMMA operands in contiguous k order, wave32) ----
// A (16x32, MxK) row-major with leading dim ld (elements):
// lane L: row = L&15 ; kb = (L>>4)*16 ; halves [kb..kb+7] and [kb+8..kb+15]
union AFragU { v16h v; v8h h[2]; };
__device__ __forceinline__ v16h load_a16x32(const _Float16* base, int ld) {
  const int lane = threadIdx.x & 31;
  const _Float16* p = base + (lane & 15) * ld + ((lane >> 4) << 4);
  AFragU f;
  f.h[0] = *(const v8h*)(p);
  f.h[1] = *(const v8h*)(p + 8);
  return f.v;
}
// B (32x16, KxN) given column-major storage: column n at colBase + n*ld,
// lane L: n = L&15 ; k-slice = (L>>4)*16 .. +15  (one contiguous 32B load)
__device__ __forceinline__ v16h load_b32x16(const _Float16* colBase, int ld) {
  const int lane = threadIdx.x & 31;
  return *(const v16h*)(colBase + (lane & 15) * ld + ((lane >> 4) << 4));
}

__device__ __forceinline__ v8f wmma_f16(v16h a, v16h b, v8f c) {
  return __builtin_amdgcn_wmma_f32_16x16x32_f16(false, a, false, b,
                                                (short)0, c, false, false);
}

// row reductions within 16-lane halves (matches C/D layout rows)
__device__ __forceinline__ float rmax16(float x) {
  x = fmaxf(x, __shfl_xor(x, 1, 32));
  x = fmaxf(x, __shfl_xor(x, 2, 32));
  x = fmaxf(x, __shfl_xor(x, 4, 32));
  x = fmaxf(x, __shfl_xor(x, 8, 32));
  return x;
}
__device__ __forceinline__ float rsum16(float x) {
  x += __shfl_xor(x, 1, 32);
  x += __shfl_xor(x, 2, 32);
  x += __shfl_xor(x, 4, 32);
  x += __shfl_xor(x, 8, 32);
  return x;
}

// ---------------- precision conversion kernels ----------------
__global__ void cvt_f32_to_f16(const float* __restrict__ in,
                               _Float16* __restrict__ out, int n) {
  int i = blockIdx.x * blockDim.x + threadIdx.x;
  if (i < n) out[i] = (_Float16)in[i];
}
// out[c*R + r] = in[r*C + c]  (in: R x C row-major, out: C x R row-major)
__global__ void transpose_cvt_f16(const float* __restrict__ in,
                                  _Float16* __restrict__ out, int R, int C) {
  int i = blockIdx.x * blockDim.x + threadIdx.x;
  if (i < R * C) {
    int r = i % R;
    int c = i / R;
    out[i] = (_Float16)in[(size_t)r * C + c];
  }
}

// ---- software-pipelined 32x64 GEMM tile core (K = 1024, ld = 1024) ----
// abase: rows m..m+31 row-major; bbase: columns n..n+63 column-major.
// acc[i][j] covers rows i*16.., cols j*16..
__device__ __forceinline__ void mm32x64_pipelined(const _Float16* __restrict__ abase,
                                                  const _Float16* __restrict__ bbase,
                                                  v8f (&acc)[2][4]) {
  v16h A0[2], B0[4], A1[2], B1[4];

  auto loadg = [&](v16h (&A)[2], v16h (&Bf)[4], int kc) {
    A[0] = load_a16x32(abase + kc, C_);
    A[1] = load_a16x32(abase + 16 * C_ + kc, C_);
#pragma unroll
    for (int j = 0; j < 4; ++j)
      Bf[j] = load_b32x16(bbase + (size_t)j * 16 * C_ + kc, C_);
  };
  auto comp = [&](const v16h (&A)[2], const v16h (&Bf)[4]) {
#pragma unroll
    for (int i = 0; i < 2; ++i)
#pragma unroll
      for (int j = 0; j < 4; ++j)
        acc[i][j] = wmma_f16(A[i], Bf[j], acc[i][j]);
  };

  loadg(A0, B0, 0);
  int kc = 0;
  // 32 k-chunks total; 2-stage ping-pong: load k+1 before computing k
  for (int it = 0; it < 15; ++it) {
    loadg(A1, B1, kc + 32);
    comp(A0, B0);
    loadg(A0, B0, kc + 64);
    comp(A1, B1);
    kc += 64;
  }
  loadg(A1, B1, 992);   // last chunk
  comp(A0, B0);         // chunk 960
  comp(A1, B1);         // chunk 992
}

// ---------------- QKV projection: xh[8192x1024] @ wT -> q,k,vT ----------------
// one wave per 32x64 tile of the 8192x3072 result; 4 waves/block share A rows
__global__ __launch_bounds__(128) void gemm_qkv_kernel(
    const _Float16* __restrict__ xh, const _Float16* __restrict__ wT,
    _Float16* __restrict__ q, _Float16* __restrict__ k,
    _Float16* __restrict__ vt) {
  const int lane = threadIdx.x & 31;
  const int wave = threadIdx.x >> 5;
  const int m32  = blockIdx.x;                // 0..255
  const int n64  = blockIdx.y * 4 + wave;     // 0..47
  const int lh = lane >> 4, ln = lane & 15;

  v8f acc[2][4];
#pragma unroll
  for (int i = 0; i < 2; ++i)
#pragma unroll
    for (int j = 0; j < 4; ++j) acc[i][j] = vzero8f();

  mm32x64_pipelined(xh + (size_t)m32 * 32 * C_,
                    wT + (size_t)n64 * 64 * C_, acc);

#pragma unroll
  for (int i = 0; i < 2; ++i) {
#pragma unroll
    for (int j = 0; j < 4; ++j) {
#pragma unroll
      for (int r = 0; r < 8; ++r) {
        const int row = m32 * 32 + i * 16 + r + 8 * lh;  // token row (0..8191)
        const int col = n64 * 64 + j * 16 + ln;          // 0..3071
        const int which = col >> 10;                     // 0=q 1=k 2=v
        const int cc = col & 1023;
        const int h = cc >> 6, d = cc & 63;
        const int bidx = row >> 11, t = row & 2047;
        const size_t bh = (size_t)(bidx * H_ + h);
        const _Float16 val = (_Float16)acc[i][j][r];
        if (which == 0)      q [(bh * T_ + t) * DK_ + d] = val;
        else if (which == 1) k [(bh * T_ + t) * DK_ + d] = val;
        else                 vt[(bh * DK_ + d) * T_ + t] = val;  // V transposed
      }
    }
  }
}

// ---------------- flash attention: one wave per (b,h,16-query tile) ----------------
__global__ __launch_bounds__(128) void attn_kernel(
    const _Float16* __restrict__ q, const _Float16* __restrict__ kmat,
    const _Float16* __restrict__ vt, _Float16* __restrict__ aout) {
  __shared__ __align__(64) _Float16 pbuf[4][16 * 32];
  const int lane = threadIdx.x & 31;
  const int wave = threadIdx.x >> 5;
  const int gw = blockIdx.x * 4 + wave;   // 0..8191
  const int qt = gw & 127;                // T/16
  const int bh = gw >> 7;                 // b*16 + h
  const int qbase = qt * 16;
  const int lh = lane >> 4, ln = lane & 15;

  const _Float16* qptr = q + ((size_t)bh * T_ + qbase) * DK_;
  const v16h aq0 = load_a16x32(qptr, DK_);        // d 0..31
  const v16h aq1 = load_a16x32(qptr + 32, DK_);   // d 32..63

  float mrow[8], lrow[8];
  v8f o[4];
#pragma unroll
  for (int r = 0; r < 8; ++r) { mrow[r] = -INFINITY; lrow[r] = 0.0f; }
#pragma unroll
  for (int c = 0; c < 4; ++c) o[c] = vzero8f();

  _Float16* myp = &pbuf[wave][0];
  const float scale = 0.125f;                 // 1/sqrt(64)
  const int nkb = (qbase + 47) >> 5;          // causal: key blocks of 32

  auto loadK = [&](v16h (&kf)[4], int keyBase) {
    const _Float16* kptr = kmat + ((size_t)bh * T_ + keyBase) * DK_;
    kf[0] = load_b32x16(kptr, DK_);                // keys +0..15, d 0..31
    kf[1] = load_b32x16(kptr + 32, DK_);           // keys +0..15, d 32..63
    kf[2] = load_b32x16(kptr + 16 * DK_, DK_);     // keys +16..31, d 0..31
    kf[3] = load_b32x16(kptr + 16 * DK_ + 32, DK_);
  };

  v16h kcur[4], knxt[4];
  loadK(kcur, 0);

  for (int kb = 0; kb < nkb; ++kb) {
    const int keyBase = kb * 32;

    // issue V loads for this block and K loads for the next block up front;
    // the softmax VALU work below hides their latency
    v16h vf[4];
    const _Float16* vbase = vt + (size_t)bh * DK_ * T_ + keyBase;
#pragma unroll
    for (int c = 0; c < 4; ++c)
      vf[c] = load_b32x16(vbase + (size_t)(c * 16) * T_, T_);
    if (kb + 1 < nkb) loadK(knxt, keyBase + 32);

    v8f s0 = vzero8f(), s1 = vzero8f();
    s0 = wmma_f16(aq0, kcur[0], s0);
    s0 = wmma_f16(aq1, kcur[1], s0);
    s1 = wmma_f16(aq0, kcur[2], s1);
    s1 = wmma_f16(aq1, kcur[3], s1);

#pragma unroll
    for (int r = 0; r < 8; ++r) {
      const int qg = qbase + r + 8 * lh;
      const int k0 = keyBase + ln;
      const float v0 = (k0 <= qg)      ? s0[r] * scale : -INFINITY;
      const float v1 = (k0 + 16 <= qg) ? s1[r] * scale : -INFINITY;
      float rm = rmax16(fmaxf(v0, v1));
      const float mn = fmaxf(mrow[r], rm);
      const float alpha = __expf(mrow[r] - mn);
      mrow[r] = mn;
      const float e0 = __expf(v0 - mn);
      const float e1 = __expf(v1 - mn);
      lrow[r] = lrow[r] * alpha + rsum16(e0 + e1);
#pragma unroll
      for (int c = 0; c < 4; ++c) o[c][r] *= alpha;
      const int row = r + 8 * lh;
      myp[row * 32 + ln]      = (_Float16)e0;   // stage P in LDS, row-major 16x32
      myp[row * 32 + 16 + ln] = (_Float16)e1;
    }
    __builtin_amdgcn_wave_barrier();
    const v16h pf = load_a16x32(myp, 32);       // reload P as A-fragment
#pragma unroll
    for (int c = 0; c < 4; ++c)
      o[c] = wmma_f16(pf, vf[c], o[c]);
    __builtin_amdgcn_wave_barrier();

    if (kb + 1 < nkb) {
#pragma unroll
      for (int i = 0; i < 4; ++i) kcur[i] = knxt[i];
    }
  }

  const int h = bh & 15, bidx = bh >> 4;
#pragma unroll
  for (int c = 0; c < 4; ++c) {
#pragma unroll
    for (int r = 0; r < 8; ++r) {
      const int row = qbase + r + 8 * lh;
      const int col = h * 64 + c * 16 + ln;
      aout[((size_t)bidx * T_ + row) * C_ + col] = (_Float16)(o[c][r] / lrow[r]);
    }
  }
}

// ---------------- output projection: attn[8192x1024] @ woutT -> f32 out ----------------
__global__ __launch_bounds__(128) void gemm_out_kernel(
    const _Float16* __restrict__ ah, const _Float16* __restrict__ wT,
    float* __restrict__ out) {
  const int lane = threadIdx.x & 31;
  const int wave = threadIdx.x >> 5;
  const int m32 = blockIdx.x;               // 0..255
  const int n64 = blockIdx.y * 4 + wave;    // 0..15
  const int lh = lane >> 4, ln = lane & 15;

  v8f acc[2][4];
#pragma unroll
  for (int i = 0; i < 2; ++i)
#pragma unroll
    for (int j = 0; j < 4; ++j) acc[i][j] = vzero8f();

  mm32x64_pipelined(ah + (size_t)m32 * 32 * C_,
                    wT + (size_t)n64 * 64 * C_, acc);

#pragma unroll
  for (int i = 0; i < 2; ++i) {
#pragma unroll
    for (int j = 0; j < 4; ++j) {
#pragma unroll
      for (int r = 0; r < 8; ++r) {
        const int row = m32 * 32 + i * 16 + r + 8 * lh;
        const int col = n64 * 64 + j * 16 + ln;
        out[(size_t)row * C_ + col] = acc[i][j][r];
      }
    }
  }
}

extern "C" void kernel_launch(void* const* d_in, const int* in_sizes, int n_in,
                              void* d_out, int out_size, void* d_ws, size_t ws_size,
                              hipStream_t stream) {
  const float* x     = (const float*)d_in[0];   // [4,2048,1024]
  const float* w_qkv = (const float*)d_in[1];   // [1024,3072]
  const float* w_out = (const float*)d_in[2];   // [1024,1024]
  float* out = (float*)d_out;

  const size_t nX   = (size_t)B_ * T_ * C_;        // 8388608
  const size_t nQKV = (size_t)C_ * 3 * C_;         // 3145728
  const size_t nWO  = (size_t)C_ * C_;             // 1048576
  const size_t nHd  = (size_t)B_ * H_ * T_ * DK_;  // 8388608

  char* ws = (char*)d_ws;
  size_t off = 0;
  _Float16* xh    = (_Float16*)(ws + off); off += nX   * sizeof(_Float16);
  _Float16* wqkvT = (_Float16*)(ws + off); off += nQKV * sizeof(_Float16);
  _Float16* woutT = (_Float16*)(ws + off); off += nWO  * sizeof(_Float16);
  _Float16* qb    = (_Float16*)(ws + off); off += nHd  * sizeof(_Float16);
  _Float16* kb    = (_Float16*)(ws + off); off += nHd  * sizeof(_Float16);
  _Float16* vtb   = (_Float16*)(ws + off); off += nHd  * sizeof(_Float16);
  _Float16* attnb = (_Float16*)(ws + off); off += nX   * sizeof(_Float16);
  (void)ws_size; (void)in_sizes; (void)n_in; (void)out_size;

  cvt_f32_to_f16<<<(int)((nX + 255) / 256), 256, 0, stream>>>(x, xh, (int)nX);
  transpose_cvt_f16<<<(int)((nQKV + 255) / 256), 256, 0, stream>>>(w_qkv, wqkvT, C_, 3 * C_);
  transpose_cvt_f16<<<(int)((nWO + 255) / 256), 256, 0, stream>>>(w_out, woutT, C_, C_);

  dim3 gQKV(256, 12);     // 256 m32-tiles x (48 n64-tiles / 4 waves)
  gemm_qkv_kernel<<<gQKV, 128, 0, stream>>>(xh, wqkvT, qb, kb, vtb);

  attn_kernel<<<2048, 128, 0, stream>>>(qb, kb, vtb, attnb);  // 8192 waves

  dim3 gOUT(256, 4);      // 16 n64-tiles / 4 waves
  gemm_out_kernel<<<gOUT, 128, 0, stream>>>(attnb, woutT, out);
}
